// MultiBoxLoss_78580721647859
// MI455X (gfx1250) — matrix operand order repaired and checked by reading, not compiled, not run on hardware
//
#include <hip/hip_runtime.h>
#include <hip/hip_bf16.h>

// Problem constants (from reference setup_inputs)
#define BB 32
#define DD 8732
#define CC 21
#define LDIM 32
#define OO 8
#define KP 256          // MAX_POS
#define TGT_STRIDE 9    // [box4, label, pose3, zero]

typedef _Float16 v16h __attribute__((ext_vector_type(16)));
typedef float    v8f  __attribute__((ext_vector_type(8)));

// ---------------------------------------------------------------------------
// Kernel 1: GT <-> prior matching (one block per batch). Reproduces match_one.
// ---------------------------------------------------------------------------
__global__ void mb_match_kernel(const float* __restrict__ tgt, const float* __restrict__ dbox,
                                int* __restrict__ conf_t, int* __restrict__ bt_idx,
                                unsigned* __restrict__ num_pos) {
  __shared__ float tb[OO][4];
  __shared__ float tarea[OO];
  __shared__ int   tlab[OO];
  __shared__ int   bp[OO];
  __shared__ float rOv[256];
  __shared__ int   rIdx[256];
  const int b = blockIdx.x, tid = threadIdx.x;
  if (tid < OO) {
    const float* tr = tgt + ((size_t)b * OO + tid) * TGT_STRIDE;
    tb[tid][0] = tr[0]; tb[tid][1] = tr[1]; tb[tid][2] = tr[2]; tb[tid][3] = tr[3];
    tarea[tid] = (tr[2] - tr[0]) * (tr[3] - tr[1]);
    tlab[tid]  = (int)tr[4];
  }
  __syncthreads();

  // Phase A: per-truth best prior (argmax over D, first-max tie-break)
  float myOv[OO]; int myIdx[OO];
  #pragma unroll
  for (int t = 0; t < OO; t++) { myOv[t] = -1.0f; myIdx[t] = 0x7fffffff; }
  for (int d = tid; d < DD; d += 256) {
    const float cx = dbox[d*4+0], cy = dbox[d*4+1], pw = dbox[d*4+2], ph = dbox[d*4+3];
    const float px0 = cx - 0.5f*pw, py0 = cy - 0.5f*ph;
    const float px1 = cx + 0.5f*pw, py1 = cy + 0.5f*ph;
    const float pa = pw * ph;
    #pragma unroll
    for (int t = 0; t < OO; t++) {
      const float ix0 = fmaxf(tb[t][0], px0), iy0 = fmaxf(tb[t][1], py0);
      const float ix1 = fminf(tb[t][2], px1), iy1 = fminf(tb[t][3], py1);
      const float iw = fmaxf(ix1 - ix0, 0.f), ih = fmaxf(iy1 - iy0, 0.f);
      const float inter = iw * ih;
      const float iou = inter / (tarea[t] + pa - inter);
      if (iou > myOv[t]) { myOv[t] = iou; myIdx[t] = d; }   // strict > keeps first max
    }
  }
  for (int t = 0; t < OO; t++) {
    rOv[tid] = myOv[t]; rIdx[tid] = myIdx[t];
    __syncthreads();
    for (int s = 128; s > 0; s >>= 1) {
      if (tid < s) {
        const float o2 = rOv[tid + s]; const int i2 = rIdx[tid + s];
        if (o2 > rOv[tid] || (o2 == rOv[tid] && i2 < rIdx[tid])) { rOv[tid] = o2; rIdx[tid] = i2; }
      }
      __syncthreads();
    }
    if (tid == 0) bp[t] = rIdx[0];
    __syncthreads();
  }

  // Phase B: per-prior best truth + override, write conf_t / bt_idx
  unsigned myPos = 0;
  for (int d = tid; d < DD; d += 256) {
    const float cx = dbox[d*4+0], cy = dbox[d*4+1], pw = dbox[d*4+2], ph = dbox[d*4+3];
    const float px0 = cx - 0.5f*pw, py0 = cy - 0.5f*ph;
    const float px1 = cx + 0.5f*pw, py1 = cy + 0.5f*ph;
    const float pa = pw * ph;
    float btOv = -1.f; int bt = 0;
    #pragma unroll
    for (int t = 0; t < OO; t++) {
      const float ix0 = fmaxf(tb[t][0], px0), iy0 = fmaxf(tb[t][1], py0);
      const float ix1 = fminf(tb[t][2], px1), iy1 = fminf(tb[t][3], py1);
      const float iw = fmaxf(ix1 - ix0, 0.f), ih = fmaxf(iy1 - iy0, 0.f);
      const float inter = iw * ih;
      const float iou = inter / (tarea[t] + pa - inter);
      if (iou > btOv) { btOv = iou; bt = t; }
    }
    #pragma unroll
    for (int t = 0; t < OO; t++)       // ascending: last write wins (JAX .at[].set dup semantics)
      if (bp[t] == d) { bt = t; btOv = 2.0f; }
    const int cf = (btOv < 0.5f) ? 0 : (tlab[bt] + 1);
    conf_t[(size_t)b * DD + d] = cf;
    bt_idx[(size_t)b * DD + d] = bt;
    myPos += (cf > 0);
  }
  rIdx[tid] = (int)myPos;
  __syncthreads();
  for (int s = 128; s > 0; s >>= 1) { if (tid < s) rIdx[tid] += rIdx[tid + s]; __syncthreads(); }
  if (tid == 0) num_pos[b] = (unsigned)rIdx[0];
}

// ---------------------------------------------------------------------------
// Kernel 2: CE per element + masked smooth-L1 loc loss + pose L2 loss.
// ---------------------------------------------------------------------------
__global__ void mb_loss_lp_ce_kernel(const float* __restrict__ loc, const float* __restrict__ conf,
                                     const float* __restrict__ pose, const float* __restrict__ tgt,
                                     const float* __restrict__ dbox,
                                     const int* __restrict__ conf_t, const int* __restrict__ bt_idx,
                                     float* __restrict__ ce, float* accF) {
  __shared__ float r1[256], r2[256];
  const int tid = threadIdx.x;
  const size_t idx = (size_t)blockIdx.x * 256 + tid;
  float ll = 0.f, lp = 0.f;
  if (idx < (size_t)BB * DD) {
    const int b = (int)(idx / DD), d = (int)(idx % DD);
    const int cf = conf_t[idx], ti = bt_idx[idx];
    const float* cr = conf + idx * CC;
    float m = cr[0];
    #pragma unroll
    for (int c = 1; c < CC; c++) m = fmaxf(m, cr[c]);
    float s = 0.f;
    #pragma unroll
    for (int c = 0; c < CC; c++) s += __expf(cr[c] - m);
    const float lse = m + __logf(s);
    ce[idx] = lse - cr[cf];
    if (cf > 0) {
      const float* tr = tgt + ((size_t)b * OO + ti) * TGT_STRIDE;
      const float pcx = dbox[d*4+0], pcy = dbox[d*4+1], pw = dbox[d*4+2], ph = dbox[d*4+3];
      float lt[4];
      lt[0] = (0.5f * (tr[0] + tr[2]) - pcx) / (0.1f * pw);
      lt[1] = (0.5f * (tr[1] + tr[3]) - pcy) / (0.1f * ph);
      lt[2] = __logf((tr[2] - tr[0]) / pw) / 0.2f;
      lt[3] = __logf((tr[3] - tr[1]) / ph) / 0.2f;
      const float* lr = loc + idx * 4;
      #pragma unroll
      for (int k2 = 0; k2 < 4; k2++) {
        const float df = lr[k2] - lt[k2]; const float a = fabsf(df);
        ll += (a < 1.f) ? 0.5f * df * df : (a - 0.5f);
      }
      const float* pr = pose + idx * 3;
      #pragma unroll
      for (int k2 = 0; k2 < 3; k2++) { const float dq = pr[k2] - tr[5 + k2]; lp += dq * dq; }
    }
  }
  r1[tid] = ll; r2[tid] = lp;
  __syncthreads();
  for (int s2 = 128; s2 > 0; s2 >>= 1) {
    if (tid < s2) { r1[tid] += r1[tid + s2]; r2[tid] += r2[tid + s2]; }
    __syncthreads();
  }
  if (tid == 0) { atomicAdd(accF + 0, r1[0]); atomicAdd(accF + 1, r2[0]); }
}

// ---------------------------------------------------------------------------
// Kernel 3: hard-negative mining. ce_mined row staged in WGP LDS (35KB << 320KB);
// stable descending rank by counting reproduces argsort(argsort(-ce_mined)).
// ---------------------------------------------------------------------------
__global__ void mb_mining_kernel(const float* __restrict__ ce, const int* __restrict__ conf_t,
                                 const unsigned* __restrict__ num_pos, float* accF) {
  __shared__ float cm[DD];
  __shared__ float rs[256];
  const int b = blockIdx.x, tid = threadIdx.x;
  const size_t base = (size_t)b * DD;
  for (int d = tid; d < DD; d += 256) {
    if (d + 256 < DD) __builtin_prefetch(ce + base + d + 256, 0, 0);
    cm[d] = (conf_t[base + d] > 0) ? 0.f : ce[base + d];
  }
  __syncthreads();
  int nn = (int)num_pos[b] * 3;
  if (nn > DD) nn = DD;
  float lc = 0.f;
  for (int d = tid; d < DD; d += 256) {
    const float v = cm[d];
    const bool isPos = conf_t[base + d] > 0;
    int rank = 0;
    for (int j = 0; j < DD; j++) {
      const float w = cm[j];
      rank += ((w > v) || (w == v && j < d)) ? 1 : 0;
    }
    if (isPos || rank < nn) lc += ce[base + d];
  }
  rs[tid] = lc;
  __syncthreads();
  for (int s = 128; s > 0; s >>= 1) { if (tid < s) rs[tid] += rs[tid + s]; __syncthreads(); }
  if (tid == 0) atomicAdd(accF + 2, rs[0]);
}

// ---------------------------------------------------------------------------
// Kernel 4: stable compaction: first 256 positive flat indices (ascending),
// then negatives -- exactly argsort(where(pos,0,1))[:256].
// ---------------------------------------------------------------------------
__global__ void mb_build_order_kernel(const int* __restrict__ conf_t,
                                      int* __restrict__ order, int* __restrict__ negbuf) {
  __shared__ int sp[256], sn[256];
  __shared__ int basePos, baseNeg;
  const int tid = threadIdx.x;
  if (tid == 0) { basePos = 0; baseNeg = 0; }
  const int total = BB * DD;
  for (int start = 0; start < total; start += 256) {
    __syncthreads();
    if (basePos >= KP && baseNeg >= KP) break;
    const int idx = start + tid;
    int isP = 0, isN = 0;
    if (idx < total) { const int p = conf_t[idx] > 0; isP = p; isN = 1 - p; }
    sp[tid] = isP; sn[tid] = isN;
    __syncthreads();
    for (int off = 1; off < 256; off <<= 1) {       // inclusive Hillis-Steele scan
      int ap = 0, an = 0;
      if (tid >= off) { ap = sp[tid - off]; an = sn[tid - off]; }
      __syncthreads();
      sp[tid] += ap; sn[tid] += an;
      __syncthreads();
    }
    const int posSlot = basePos + sp[tid] - isP;
    const int negSlot = baseNeg + sn[tid] - isN;
    if (isP && posSlot < KP) order[posSlot] = idx;
    if (isN && negSlot < KP) negbuf[negSlot] = idx;
    __syncthreads();
    if (tid == 255) { basePos += sp[255]; baseNeg += sn[255]; }
  }
  __syncthreads();
  const int tp = basePos;
  if (tp < KP)
    for (int s = tp + tid; s < KP; s += 256) order[s] = negbuf[s - tp];
}

// ---------------------------------------------------------------------------
// Kernel 5: gather + normalize (wave32 shfl reductions), emit f16 operands and
// exact f32 squared norms of the f16-rounded values for the || ||^2 terms.
// ---------------------------------------------------------------------------
__global__ void mb_gather_kernel(const float* __restrict__ line, const float* __restrict__ tgt,
                                 const int* __restrict__ conf_t, const int* __restrict__ bt_idx,
                                 const int* __restrict__ order,
                                 _Float16* __restrict__ emb16, _Float16* __restrict__ nemb16,
                                 float* __restrict__ e2, float* __restrict__ n2,
                                 int* __restrict__ lab, int* __restrict__ valid,
                                 float* __restrict__ qp) {
  const int k = blockIdx.x, t = threadIdx.x;   // 32 threads = 1 wave
  const int flat = order[k];
  const float e = line[(size_t)flat * LDIM + t];
  float s = e * e;
  #pragma unroll
  for (int m = 16; m > 0; m >>= 1) s += __shfl_xor(s, m, 32);
  const float nv = e / fmaxf(sqrtf(s), 1e-12f);
  const _Float16 eh = (_Float16)e;
  const _Float16 nh = (_Float16)nv;
  emb16[k * LDIM + t] = eh;
  nemb16[k * LDIM + t] = nh;
  float se = (float)eh * (float)eh;
  float sn = (float)nh * (float)nh;
  #pragma unroll
  for (int m = 16; m > 0; m >>= 1) { se += __shfl_xor(se, m, 32); sn += __shfl_xor(sn, m, 32); }
  if (t == 0) {
    e2[k] = se; n2[k] = sn;
    const int cf = conf_t[flat];
    lab[k] = cf; valid[k] = (cf > 0);
    const int bb = flat / DD;
    const int ti = bt_idx[flat];
    const float* tr = tgt + ((size_t)bb * OO + ti) * TGT_STRIDE;
    qp[k*3+0] = tr[5]; qp[k*3+1] = tr[6]; qp[k*3+2] = tr[7];
  }
}

// ---------------------------------------------------------------------------
// Kernel 6: WMMA Gram matrices. One wave per 16x16 tile; K=32 == embedding dim
// so each Gram tile is exactly one v_wmma_f32_16x16x32_f16.
// Fragment packing per ISA 7.12.2 (wave32 layouts).
// ---------------------------------------------------------------------------
__global__ void mb_gram_wmma_kernel(const _Float16* __restrict__ emb16,
                                    const _Float16* __restrict__ nemb16,
                                    const float* __restrict__ e2, const float* __restrict__ n2,
                                    float* __restrict__ dist, float* __restrict__ esq) {
  const int tj = blockIdx.x, ti = blockIdx.y;
  const int lane = threadIdx.x;
  const int half = lane >> 4;
  const int lm = lane & 15;
  union { v16h v; _Float16 h[16]; } aN, bN, aE, bE;
  const int rowA = ti * 16 + lm;   // A-matrix row M
  const int rowB = tj * 16 + lm;   // B column N == row of E (B = E^T)
  #pragma unroll
  for (int v = 0; v < 8; v++) {
    // A 16x32 f16 layout: V0..3 hold K = half*8 + {0..7}; V4..7 hold K = 16 + half*8 + {0..7}
    const int kA = (v < 4) ? (half * 8 + 2 * v) : (16 + half * 8 + 2 * (v - 4));
    // B 32x16 f16 layout: V0..3 hold K = half*8 + {0..7}; V4..7 hold K = 16 + half*8 + {0..7}
    const int kB = (v < 4) ? (half * 8 + 2 * v) : (16 + half * 8 + 2 * (v - 4));
    aN.h[2*v]   = nemb16[rowA * LDIM + kA];
    aN.h[2*v+1] = nemb16[rowA * LDIM + kA + 1];
    bN.h[2*v]   = nemb16[rowB * LDIM + kB];
    bN.h[2*v+1] = nemb16[rowB * LDIM + kB + 1];
    aE.h[2*v]   = emb16[rowA * LDIM + kA];
    aE.h[2*v+1] = emb16[rowA * LDIM + kA + 1];
    bE.h[2*v]   = emb16[rowB * LDIM + kB];
    bE.h[2*v+1] = emb16[rowB * LDIM + kB + 1];
  }
  union { v8f v; float f[8]; } cN, cE;
  #pragma unroll
  for (int r = 0; r < 8; r++) { cN.f[r] = 0.f; cE.f[r] = 0.f; }
  cN.v = __builtin_amdgcn_wmma_f32_16x16x32_f16(false, aN.v, false, bN.v, (short)0, cN.v, false, false);
  cE.v = __builtin_amdgcn_wmma_f32_16x16x32_f16(false, aE.v, false, bE.v, (short)0, cE.v, false, false);
  #pragma unroll
  for (int r = 0; r < 8; r++) {
    const int i = ti * 16 + r + half * 8;   // C/D layout: VGPR r -> M=r (lanes 0-15) / M=8+r
    const int j = tj * 16 + lm;             // N = lane & 15
    const float sqn = n2[i] + n2[j] - 2.f * cN.f[r];
    dist[i * KP + j] = sqrtf(fmaxf(sqn, 1e-12f));
    esq[i * KP + j]  = e2[i] + e2[j] - 2.f * cE.f[r];
  }
}

// ---------------------------------------------------------------------------
// Kernel 7: triplet + desk terms. Block = anchor row i; dist row + neg flags in LDS.
// ---------------------------------------------------------------------------
__global__ void mb_triplet_kernel(const float* __restrict__ dist, const float* __restrict__ esq,
                                  const int* __restrict__ lab, const int* __restrict__ valid,
                                  const float* __restrict__ qp,
                                  float* accF, unsigned* accU) {
  __shared__ float drow[KP];
  __shared__ unsigned char npf[KP];
  __shared__ float rf1[256], rf2[256];
  __shared__ unsigned ru1[256], ru2[256], ru3[256];
  const int i = blockIdx.x, t = threadIdx.x;
  const int vi = valid[i], li = lab[i];
  drow[t] = dist[i * KP + t];
  const int vt = valid[t], lt2 = lab[t];
  __syncthreads();
  const bool np = vi && vt && (t != i) && (li != lt2) && (drow[t] < 0.8f);
  const bool pp = vi && vt && (t != i) && (li == lt2) && (drow[t] > 0.2f);
  npf[t] = np ? 1 : 0;
  __syncthreads();
  float desk = 0.f, tsum = 0.f; unsigned tnz = 0;
  if (pp) {
    const float q0 = qp[i*3+0] - qp[t*3+0];
    const float q1 = qp[i*3+1] - qp[t*3+1];
    const float q2 = qp[i*3+2] - qp[t*3+2];
    const float qsq = q0*q0 + q1*q1 + q2*q2;
    const float ev = esq[i * KP + t] - qsq;
    desk = ev * ev;
    const float dij = drow[t];
    for (int k2 = 0; k2 < KP; k2++) {
      if (npf[k2]) {
        const float l = dij - drow[k2] + 0.2f;
        if (l > 0.f) { tsum += l; tnz++; }
      }
    }
  }
  rf1[t] = tsum; rf2[t] = desk;
  ru1[t] = pp ? 1u : 0u; ru2[t] = np ? 1u : 0u; ru3[t] = tnz;
  __syncthreads();
  for (int s = 128; s > 0; s >>= 1) {
    if (t < s) {
      rf1[t] += rf1[t+s]; rf2[t] += rf2[t+s];
      ru1[t] += ru1[t+s]; ru2[t] += ru2[t+s]; ru3[t] += ru3[t+s];
    }
    __syncthreads();
  }
  if (t == 0) {
    atomicAdd(accF + 3, rf1[0]);    // trip sum (already relu'd, >0 only)
    atomicAdd(accF + 4, rf2[0]);    // desk sum
    atomicAdd(accU + 0, ru1[0]);    // n_pp
    atomicAdd(accU + 1, ru2[0]);    // n_np
    atomicAdd(accU + 2, ru3[0]);    // nz count
  }
}

// ---------------------------------------------------------------------------
// Kernel 8: finalize the 5 scalars.
// ---------------------------------------------------------------------------
__global__ void mb_finalize_kernel(const float* __restrict__ accF, const unsigned* __restrict__ accU,
                                   const unsigned* __restrict__ num_pos, float* __restrict__ out) {
  unsigned sp = 0;
  for (int b2 = 0; b2 < BB; b2++) sp += num_pos[b2];
  const float N = fmaxf((float)sp, 1.0f);
  const float loss_l = accF[0] / N;
  const float loss_p = accF[1] / N;
  const float loss_c = accF[2] / N;
  const unsigned npp = accU[0], nnp = accU[1], nz = accU[2];
  const float loss_t = accF[3] / (float)(nz > 0u ? nz : 1u);
  const float nppf = fmaxf((float)npp, 1.0f);
  float Ldesk = accF[4] / nppf + loss_t / (float)((npp + nnp) > 0u ? (npp + nnp) : 1u);
  Ldesk = Ldesk / nppf / 32.0f;
  out[0] = loss_l; out[1] = loss_c; out[2] = loss_p; out[3] = Ldesk; out[4] = loss_t;
}

// ---------------------------------------------------------------------------
extern "C" void kernel_launch(void* const* d_in, const int* in_sizes, int n_in,
                              void* d_out, int out_size, void* d_ws, size_t ws_size,
                              hipStream_t stream) {
  (void)in_sizes; (void)n_in; (void)out_size; (void)ws_size;
  const float* loc  = (const float*)d_in[0];
  const float* conf = (const float*)d_in[1];
  const float* line = (const float*)d_in[2];
  const float* pose = (const float*)d_in[3];
  const float* dbox = (const float*)d_in[4];
  const float* tgt  = (const float*)d_in[5];
  float* out = (float*)d_out;

  char* w = (char*)d_ws;
  size_t off = 0;
  auto wsalloc = [&](size_t bytes) -> void* {
    void* p = (void*)(w + off);
    off += (bytes + 255) & ~(size_t)255;
    return p;
  };
  int*      conf_t  = (int*)wsalloc((size_t)BB * DD * sizeof(int));
  int*      bt_idx  = (int*)wsalloc((size_t)BB * DD * sizeof(int));
  float*    ce      = (float*)wsalloc((size_t)BB * DD * sizeof(float));
  int*      order   = (int*)wsalloc(KP * sizeof(int));
  int*      negbuf  = (int*)wsalloc(KP * sizeof(int));
  int*      lab     = (int*)wsalloc(KP * sizeof(int));
  int*      valid   = (int*)wsalloc(KP * sizeof(int));
  float*    qp      = (float*)wsalloc(KP * 3 * sizeof(float));
  float*    e2      = (float*)wsalloc(KP * sizeof(float));
  float*    n2      = (float*)wsalloc(KP * sizeof(float));
  _Float16* emb16   = (_Float16*)wsalloc((size_t)KP * LDIM * sizeof(_Float16));
  _Float16* nemb16  = (_Float16*)wsalloc((size_t)KP * LDIM * sizeof(_Float16));
  float*    dist    = (float*)wsalloc((size_t)KP * KP * sizeof(float));
  float*    esq     = (float*)wsalloc((size_t)KP * KP * sizeof(float));
  unsigned* num_pos = (unsigned*)wsalloc(BB * sizeof(unsigned));
  float*    accF    = (float*)wsalloc(256);           // [0..4] float accumulators
  unsigned* accU    = (unsigned*)(accF + 8);          // [0..2] count accumulators

  hipMemsetAsync(accF, 0, 64, stream);  // zero float+uint accumulators each call

  mb_match_kernel<<<BB, 256, 0, stream>>>(tgt, dbox, conf_t, bt_idx, num_pos);

  const int nblk = (BB * DD + 255) / 256;
  mb_loss_lp_ce_kernel<<<nblk, 256, 0, stream>>>(loc, conf, pose, tgt, dbox,
                                                 conf_t, bt_idx, ce, accF);

  mb_mining_kernel<<<BB, 256, 0, stream>>>(ce, conf_t, num_pos, accF);

  mb_build_order_kernel<<<1, 256, 0, stream>>>(conf_t, order, negbuf);

  mb_gather_kernel<<<KP, 32, 0, stream>>>(line, tgt, conf_t, bt_idx, order,
                                          emb16, nemb16, e2, n2, lab, valid, qp);

  mb_gram_wmma_kernel<<<dim3(16, 16), 32, 0, stream>>>(emb16, nemb16, e2, n2, dist, esq);

  mb_triplet_kernel<<<KP, 256, 0, stream>>>(dist, esq, lab, valid, qp, accF, accU);

  mb_finalize_kernel<<<1, 1, 0, stream>>>(accF, accU, num_pos, out);
}
